// GRUD_55490977464474
// MI455X (gfx1250) — matrix operand order repaired and checked by reading, not compiled, not run on hardware
//
#include <hip/hip_runtime.h>

// ---------------- CDNA5 (gfx1250) GRU-D persistent-recurrence kernel ----------------
// B=256, S=1024, I=64, H=128, NC=10. Gates G=4H=512, fused GEMM K = 2I+H = 256.
// grid = 16 workgroups (one 16-row batch tile each), block = 512 threads = 16 waves.
// Weights pinned in VGPRs as bf16 WMMA B-fragments; recurrence entirely in LDS/regs.
// Step t+1 global loads issued during step t and NOT flushed at barriers: all
// inter-phase dependencies are LDS-only, so barriers are s_wait_dscnt-scoped.

typedef __attribute__((ext_vector_type(16))) __bf16 v16bf;
typedef __attribute__((ext_vector_type(2)))  __bf16 v2bf;
typedef __attribute__((ext_vector_type(8)))  float  v8f;

union BF16x16 { v16bf v; __bf16 s[16]; };

// LDS-scoped workgroup barrier: wait only DScnt (LDS) before signaling, so
// in-flight global loads (next-step pipeline) and stores (impute stream) are
// NOT serialized the way __syncthreads()'s full fence would.
static __device__ __forceinline__ void lds_barrier() {
    asm volatile("s_wait_dscnt 0x0\n\t"
                 "s_barrier_signal -1\n\t"
                 "s_barrier_wait -1" ::: "memory");
}

// A-fragment (16x32 bf16) per-lane load from an LDS row (row-major, padded stride).
// Contiguous k order (matches B): lane<16 -> M=lane, K=k0..k0+15 with k0=kt*32
//             lane>=16 -> same M, K shifted by +16.
static __device__ __forceinline__ v16bf load_a_frag(const __bf16* rowbase, int k0) {
    union { v16bf v; uint4 q[2]; } u;
    u.q[0] = *reinterpret_cast<const uint4*>(rowbase + k0);
    u.q[1] = *reinterpret_cast<const uint4*>(rowbase + k0 + 8);
    return u.v;
}

static __device__ __forceinline__ v8f wmma_bf16(v16bf a, v16bf b, v8f c) {
    // (neg_a, A, neg_b, B, c_mod, C, reuse_a, reuse_b)
    return __builtin_amdgcn_wmma_f32_16x16x32_bf16(false, a, false, b, (short)0, c, false, false);
}

union GatesAlias {                    // d_bf is dead once gamma_h GEMM consumed it;
    float  gates[16][516];            // gates region reuses the space (barrier-separated).
    __bf16 dbf[16][72];
};

#define kS  1024
#define kI  64
#define kH  128
#define kNC 10

__global__ __launch_bounds__(512, 1)
void grud_persistent(const float* __restrict__ values, const int* __restrict__ masks,
                     const float* __restrict__ deltas,
                     const float* __restrict__ W_dh, const float* __restrict__ b_dh,
                     const float* __restrict__ W_dx, const float* __restrict__ b_dx,
                     const float* __restrict__ W_ih, const float* __restrict__ W_hh,
                     const float* __restrict__ b_ih, const float* __restrict__ b_hh,
                     const float* __restrict__ W_out, const float* __restrict__ b_out,
                     float* __restrict__ out) {
    // A-matrix staging: [x_h(64) | m(64) | h*gamma_h(128)] bf16, padded row (264) for banked b128 loads
    __shared__ __align__(16) __bf16 As[16][264];
    __shared__ __align__(16) GatesAlias sh;
    __shared__ __align__(16) float h_lds[16][132];           // raw h carried across steps
    __shared__ float wdx_s[kI], bdx_s[kI];

    const int tid  = threadIdx.x;
    const int w    = tid >> 5;        // wave id 0..15 (owns gate cols [32w, 32w+32))
    const int lane = tid & 31;
    const int nlo  = lane & 15;
    const int hi   = lane >> 4;
    const int bg   = blockIdx.x;      // batch tile: rows 16*bg .. 16*bg+15

    float c_reg[4] = {0.f, 0.f, 0.f, 0.f};   // LSTM cell state: thread owns (row tid>>5, cols (tid&31)*4+e)

    for (int k = tid; k < 16 * 132; k += 512) ((float*)h_lds)[k] = 0.f;
    if (tid < kI) { wdx_s[tid] = W_dx[tid * kI + tid]; bdx_s[tid] = b_dx[tid]; }

    // Fused gate bias (b_ih + b_hh) for this lane's two output columns
    float bias_r[2];
#pragma unroll
    for (int s = 0; s < 2; ++s) {
        int col = (2 * w + s) * 16 + nlo;
        bias_r[s] = b_ih[col] + b_hh[col];
    }

    // ---- Pin gate weights in VGPRs as bf16 B-fragments (K,N)=(32,16) per (s,kt) ----
    // B layout: lane<16 -> N=lane, K=k0..k0+15 ; lane>=16 -> N=lane-16, K=k0+16..k0+31
    v16bf bw[2][8];
#pragma unroll
    for (int s = 0; s < 2; ++s) {
        const int g = (2 * w + s) * 16 + nlo;   // gate column (row of W_ih / W_hh)
#pragma unroll
        for (int kt = 0; kt < 8; ++kt) {
            const int k0 = kt * 32 + hi * 16;
            const float* src = (k0 < 2 * kI) ? (W_ih + g * (2 * kI) + k0)
                                             : (W_hh + g * kH + (k0 - 2 * kI));
            BF16x16 u;
#pragma unroll
            for (int j = 0; j < 16; ++j) u.s[j] = (__bf16)src[j];
            bw[s][kt] = u.v;
        }
    }

    // gamma_h decay weights: waves 0..7 own 16 of the 128 hidden columns
    v16bf bdhf[2];
    float bdh_r = 0.f;
    if (w < 8) {
        const int cdh = w * 16 + nlo;
        bdh_r = b_dh[cdh];
#pragma unroll
        for (int kt = 0; kt < 2; ++kt) {
            const int k0 = kt * 32 + hi * 16;
            const float* src = W_dh + cdh * kI + k0;
            BF16x16 u;
#pragma unroll
            for (int j = 0; j < 16; ++j) u.s[j] = (__bf16)src[j];
            bdhf[kt] = u.v;
        }
    } else {
        BF16x16 z;
#pragma unroll
        for (int j = 0; j < 16; ++j) z.s[j] = (__bf16)0.f;
        bdhf[0] = z.v; bdhf[1] = z.v;
    }
    lds_barrier();

    // Software-pipelined input stream: preload step 0
    const int row  = tid >> 5;
    const int col2 = (tid & 31) * 2;
    const long rowbase = (long)(bg * 16 + row) * kS * kI + col2;
    float2 xv = *(const float2*)(values + rowbase);
    int2   mv = *(const int2*)(masks + rowbase);
    float2 dv = *(const float2*)(deltas + rowbase);

    // =========================== time recurrence ===========================
    for (int t = 0; t < kS; ++t) {
        const long base = rowbase + (long)t * kI;
        // ---- phase 0: input decay + imputation from preloaded regs; stage A[:,0:128], d_bf ----
        {
            float xh[2];
#pragma unroll
            for (int j = 0; j < 2; ++j) {
                const float x  = j ? xv.y : xv.x;
                const float d  = j ? dv.y : dv.x;
                const float mf = (float)(j ? mv.y : mv.x);
                const int   cc = col2 + j;
                const float gx = __expf(-fmaxf(fmaf(d, wdx_s[cc], bdx_s[cc]), 0.f));
                xh[j] = mf * x + (1.f - mf) * (1.f - gx);
            }
            v2bf p;
            p.x = (__bf16)xh[0];            p.y = (__bf16)xh[1];
            *(v2bf*)&As[row][col2] = p;                         // packed b32 stores
            p.x = (__bf16)(float)mv.x;      p.y = (__bf16)(float)mv.y;
            *(v2bf*)&As[row][kI + col2] = p;
            p.x = (__bf16)dv.x;             p.y = (__bf16)dv.y;
            *(v2bf*)&sh.dbf[row][col2] = p;
            float2 st; st.x = xh[0]; st.y = xh[1];
            *(float2*)(out + 2560 + base) = st;                 // impute_tensor output (drains in background)

            // issue step t+1 loads now; they stay in flight across all 4 barriers
            if (t + 1 < kS) {
                xv = *(const float2*)(values + base + kI);
                mv = *(const int2*)(masks + base + kI);
                dv = *(const float2*)(deltas + base + kI);
            }
        }
        lds_barrier();

        // ---- phase 1 (waves 0-7): gamma_h = exp(-relu(d@W_dh^T+b)), stage A[:,128:256] = bf16(h*gamma_h) ----
        if (w < 8) {
            v8f acc = {0.f, 0.f, 0.f, 0.f, 0.f, 0.f, 0.f, 0.f};
            const __bf16* arow = &sh.dbf[nlo][0];
#pragma unroll
            for (int kt = 0; kt < 2; ++kt) {
                v16bf a = load_a_frag(arow, kt * 32 + hi * 16);
                acc = wmma_bf16(a, bdhf[kt], acc);
            }
            const int col = w * 16 + nlo;
#pragma unroll
            for (int r = 0; r < 8; ++r) {
                const int M  = r + hi * 8;
                const float gh = __expf(-fmaxf(acc[r] + bdh_r, 0.f));
                As[M][2 * kI + col] = (__bf16)(h_lds[M][col] * gh);
            }
        }
        lds_barrier();

        // ---- phase 2: gates(16,512) = A(16,256) @ Wcat(256,512) + bias  (16 WMMA / wave) ----
        {
            v8f acc0 = {0.f, 0.f, 0.f, 0.f, 0.f, 0.f, 0.f, 0.f};
            v8f acc1 = {0.f, 0.f, 0.f, 0.f, 0.f, 0.f, 0.f, 0.f};
            const __bf16* arow = &As[nlo][0];
#pragma unroll
            for (int kt = 0; kt < 8; ++kt) {
                v16bf a = load_a_frag(arow, kt * 32 + hi * 16);
                acc0 = wmma_bf16(a, bw[0][kt], acc0);
                acc1 = wmma_bf16(a, bw[1][kt], acc1);
            }
#pragma unroll
            for (int r = 0; r < 8; ++r) {
                const int M = r + hi * 8;
                sh.gates[M][(2 * w + 0) * 16 + nlo] = acc0[r] + bias_r[0];
                sh.gates[M][(2 * w + 1) * 16 + nlo] = acc1[r] + bias_r[1];
            }
        }
        lds_barrier();

        // ---- phase 3: LSTM cell update; c in regs, new h -> LDS ----
        {
            const int m  = tid >> 5;
            const int j0 = (tid & 31) * 4;
#pragma unroll
            for (int e = 0; e < 4; ++e) {
                const int j = j0 + e;
                const float gi = sh.gates[m][j];
                const float gf = sh.gates[m][kH + j];
                const float gg = sh.gates[m][2 * kH + j];
                const float go = sh.gates[m][3 * kH + j];
                const float si = 1.f / (1.f + __expf(-gi));
                const float sf = 1.f / (1.f + __expf(-gf));
                const float so = 1.f / (1.f + __expf(-go));
                const float tg = tanhf(gg);
                c_reg[e] = sf * c_reg[e] + si * tg;
                h_lds[m][j] = so * tanhf(c_reg[e]);
            }
        }
        lds_barrier();
    }

    // ---- classification head: y = h @ W_out^T + b_out  (tiny: 160 dots of length 128) ----
    if (tid < 16 * kNC) {
        const int r2 = tid / kNC;
        const int nc = tid - r2 * kNC;
        float sum = b_out[nc];
        for (int k = 0; k < kH; ++k) sum = fmaf(h_lds[r2][k], W_out[nc * kH + k], sum);
        out[(long)(bg * 16 + r2) * kNC + nc] = sum;
    }
}

extern "C" void kernel_launch(void* const* d_in, const int* in_sizes, int n_in,
                              void* d_out, int out_size, void* d_ws, size_t ws_size,
                              hipStream_t stream) {
    (void)in_sizes; (void)n_in; (void)d_ws; (void)ws_size; (void)out_size;
    const float* values = (const float*)d_in[0];
    const int*   masks  = (const int*)  d_in[1];
    const float* deltas = (const float*)d_in[2];
    const float* W_dh   = (const float*)d_in[3];
    const float* b_dh   = (const float*)d_in[4];
    const float* W_dx   = (const float*)d_in[5];
    const float* b_dx   = (const float*)d_in[6];
    const float* W_ih   = (const float*)d_in[7];
    const float* W_hh   = (const float*)d_in[8];
    const float* b_ih   = (const float*)d_in[9];
    const float* b_hh   = (const float*)d_in[10];
    const float* W_out  = (const float*)d_in[11];
    const float* b_out  = (const float*)d_in[12];
    float* out = (float*)d_out;

    grud_persistent<<<dim3(16), dim3(512), 0, stream>>>(
        values, masks, deltas, W_dh, b_dh, W_dx, b_dx,
        W_ih, W_hh, b_ih, b_hh, W_out, b_out, out);
}
